// WindowDecoder_59596966199351
// MI455X (gfx1250) — matrix operand order repaired and checked by reading, not compiled, not run on hardware
//
#include <hip/hip_runtime.h>

typedef __attribute__((ext_vector_type(16))) __bf16 v16bf;
typedef __attribute__((ext_vector_type(8)))  __bf16 v8bf;
typedef __attribute__((ext_vector_type(8)))  float  v8f;

#define EMB  256
#define HID  512
#define OUTD 64
#define TT   128
#define BWROWS 512

__device__ __forceinline__ __bf16 f2bf(float f) {
  unsigned u = __float_as_uint(f);
  u += 0x7FFFu + ((u >> 16) & 1u);           // round-to-nearest-even
  unsigned short s = (unsigned short)(u >> 16);
  __bf16 r; __builtin_memcpy(&r, &s, 2); return r;
}

// gfx1250 hardware tanh (TRANS32). v_nop covers the trans-op hazard window
// (inline asm bypasses the compiler's hazard insertion).
__device__ __forceinline__ float tanh_fast(float x) {
  float r;
  asm("v_tanh_f32 %0, %1\n\tv_nop" : "=v"(r) : "v"(x));
  return r;
}

// branch-free fast sigmoid: v_exp_f32 + v_rcp_f32 (no IEEE divide expansion,
// no EXEC manipulation between WMMAs)
__device__ __forceinline__ float sigmoid_fast(float x) {
  return __builtin_amdgcn_rcpf(1.0f + __expf(-x));
}

__device__ __forceinline__ float eluf_(float x) { return x > 0.f ? x : (__expf(x) - 1.f); }

__device__ __forceinline__ v16bf make16(v8bf lo, v8bf hi) {
  return __builtin_shufflevector(lo, hi, 0,1,2,3,4,5,6,7,8,9,10,11,12,13,14,15);
}

// A fragment (16xK=32) for v_wmma_f32_16x16x32_bf16 from a row-major bf16 tile.
// base = &tile[row0][k0], ldk = row stride in elements.
// lane<16: M=lane, K = {0..7, 16..23}; lane>=16: M=lane-16, K = {8..15, 24..31}.
__device__ __forceinline__ v16bf load_a(const __bf16* tile, int ldk, int lane) {
  int m  = lane & 15;
  int kh = (lane >> 4) << 3;                 // 0 or 8
  const __bf16* p = tile + m * ldk + kh;
  v8bf lo = *(const v8bf*)(p);
  v8bf hi = *(const v8bf*)(p + 16);
  return make16(lo, hi);
}

// B fragment (K=32 x 16): B[k][n] = W[nb+n][kb+k] with W row-major (N x ldk).
// lane<16: N=lane, K=kb+{0..7, 16..23} ; lane>=16: N=lane-16, K=kb+{8..15, 24..31}.
// -> two 16-byte runs of a weight row per lane.
__device__ __forceinline__ v16bf load_b(const __bf16* W, int ldk, int nb, int kb, int lane) {
  int n = nb + (lane & 15);
  int k = kb + ((lane >> 4) << 3);
  const __bf16* p = W + n * ldk + k;
  v8bf lo = *(const v8bf*)(p);
  v8bf hi = *(const v8bf*)(p + 16);
  return make16(lo, hi);
}

__device__ __forceinline__ v8f wmma_bf16(v16bf a, v16bf b, v8f c) {
  return __builtin_amdgcn_wmma_f32_16x16x32_bf16(false, a, false, b, (short)0, c, false, false);
}

// ---------------------------------------------------------------- prep kernels
__global__ void k_convert(const float* __restrict__ s, __bf16* __restrict__ d, int n) {
  int i = blockIdx.x * 256 + threadIdx.x;
  if (i < n) d[i] = f2bf(s[i]);
}

__global__ void k_embed(const float* __restrict__ z_t, const float* __restrict__ z_g,
                        const float* __restrict__ w_emb, float* __restrict__ h0) {
  int row = blockIdx.x;        // 0..511 = b*16 + w
  int col = threadIdx.x;       // 0..255
  int b = row >> 4;
  const float* zt = z_t + row * 64;
  const float* zg = z_g + b * 64;
  const float* we = w_emb + col * 128;
  float acc = 0.f;
  #pragma unroll 8
  for (int i = 0; i < 64; ++i) acc += zt[i] * we[i];
  #pragma unroll 8
  for (int i = 0; i < 64; ++i) acc += zg[i] * we[64 + i];
  h0[row * EMB + col] = acc;
}

// ---------------------------------------------------------------- GRU recurrence
// grid = 32 WGs x 256 threads. WG owns one 16-row M-tile for all 128 steps.
// Wave q (0..7) owns a 32-column slice of the hidden state and computes all
// three gates for it: 2 N-tiles x 3 gates x 8 K-steps = 48 WMMA per step.
// A fragments are re-read from the 8 KB LDS h-tile right before each WMMA
// (no private fragment arrays -> no scratch spills).
__global__ void __launch_bounds__(256)
k_gru(const float* __restrict__ h0,
      const __bf16* __restrict__ whh,      // (768,256) bf16 row-major
      const float* __restrict__ bih, const float* __restrict__ bhh,
      __bf16* __restrict__ hs)             // (BW, T, EMB) bf16
{
  __shared__ __align__(16) __bf16 hlds[16 * EMB];         // 8 KB
  const int lane = threadIdx.x & 31;
  const int wave = threadIdx.x >> 5;
  const int rowbase = blockIdx.x * 16;
  const int colbase = wave * 32;
  const int hi8 = (lane >> 4) << 3;
  const int nlo = lane & 15;

  float bi[3][2], bh[3][2];                 // per-lane gate biases (C-layout: N = lane&15)
  #pragma unroll
  for (int g = 0; g < 3; ++g)
    #pragma unroll
    for (int j = 0; j < 2; ++j) {
      int idx = g * EMB + colbase + 16 * j + nlo;
      bi[g][j] = bih[idx]; bh[g][j] = bhh[idx];
    }

  v8f h[2];                                 // this wave's 16x32 slice of h (C-layout, f32)
  #pragma unroll
  for (int j = 0; j < 2; ++j)
    #pragma unroll
    for (int r = 0; r < 8; ++r)
      h[j][r] = h0[(rowbase + hi8 + r) * EMB + colbase + 16 * j + nlo];

  for (int t = 0; t < TT; ++t) {
    // publish current h (bf16) for A-fragment formation
    #pragma unroll
    for (int j = 0; j < 2; ++j)
      #pragma unroll
      for (int r = 0; r < 8; ++r)
        hlds[(hi8 + r) * EMB + colbase + 16 * j + nlo] = f2bf(h[j][r]);
    __syncthreads();

    #pragma unroll
    for (int j = 0; j < 2; ++j) {
      const int nb = colbase + 16 * j;
      v8f ar, az, an;
      #pragma unroll
      for (int r = 0; r < 8; ++r) { ar[r] = bh[0][j]; az[r] = bh[1][j]; an[r] = bh[2][j]; }
      #pragma unroll 1
      for (int kk = 0; kk < 8; ++kk) {
        v16bf a = load_a(hlds + kk * 32, EMB, lane);     // ds_load_b128 x2
        ar = wmma_bf16(a, load_b(whh, EMB, 0 * EMB + nb, kk * 32, lane), ar);
        az = wmma_bf16(a, load_b(whh, EMB, 1 * EMB + nb, kk * 32, lane), az);
        an = wmma_bf16(a, load_b(whh, EMB, 2 * EMB + nb, kk * 32, lane), an);
      }
      #pragma unroll
      for (int r = 0; r < 8; ++r) {
        float rr = sigmoid_fast(bi[0][j] + ar[r]);
        float zz = sigmoid_fast(bi[1][j] + az[r]);
        float nn = tanh_fast(bi[2][j] + rr * an[r]);
        h[j][r] = (1.f - zz) * nn + zz * h[j][r];
      }
      #pragma unroll
      for (int r = 0; r < 8; ++r)
        hs[((rowbase + hi8 + r) * TT + t) * EMB + nb + nlo] = f2bf(h[j][r]);
    }
    __syncthreads();                        // all A reads done before next publish
  }
}

// ---------------------------------------------------------------- fused 4-layer MLP
// grid = 512 WGs x 256 threads; wave owns a 16-row tile; barrier-free per-wave
// ping-pong act buffers in dynamic LDS (8 waves * 2 * 16x512 bf16 = 256 KB,
// within the 320 KB WGP limit). A fragments are re-read per WMMA.
__global__ void __launch_bounds__(256)
k_mlp(const __bf16* __restrict__ hs,
      const __bf16* __restrict__ w1, const float* __restrict__ b1,
      const __bf16* __restrict__ w2, const float* __restrict__ b2,
      const __bf16* __restrict__ w3, const float* __restrict__ b3,
      const __bf16* __restrict__ wmu, const float* __restrict__ bmu,
      float* __restrict__ out)
{
  extern __shared__ char smem_raw[];
  __bf16* smem = (__bf16*)smem_raw;
  const int lane = threadIdx.x & 31;
  const int wave = threadIdx.x >> 5;
  const int rowbase = blockIdx.x * 128 + wave * 16;
  const int hi8 = (lane >> 4) << 3;
  const int nlo = lane & 15;
  __bf16* a0 = smem + wave * 2 * 16 * HID;
  __bf16* a1 = a0 + 16 * HID;

  // stage this wave's 16x256 hs tile into LDS once (read 32x by layer 1)
  {
    const v8bf* g = (const v8bf*)(hs + rowbase * EMB);
    v8bf* l = (v8bf*)a1;
    #pragma unroll 1
    for (int i = lane; i < 16 * EMB / 8; i += 32) l[i] = g[i];
  }

  // Layer 1: (16x256 in a1) @ w1^T -> ELU -> a0 (16x512)
  #pragma unroll 1
  for (int n = 0; n < HID / 16; ++n) {
    float bv = b1[n * 16 + nlo];
    v8f acc;
    #pragma unroll
    for (int r = 0; r < 8; ++r) acc[r] = bv;
    #pragma unroll 1
    for (int kk = 0; kk < 8; ++kk)
      acc = wmma_bf16(load_a(a1 + kk * 32, EMB, lane),
                      load_b(w1, EMB, n * 16, kk * 32, lane), acc);
    #pragma unroll
    for (int r = 0; r < 8; ++r) a0[(hi8 + r) * HID + n * 16 + nlo] = f2bf(eluf_(acc[r]));
  }

  // Layer 2: a0 -> ELU -> a1
  #pragma unroll 1
  for (int n = 0; n < HID / 16; ++n) {
    float bv = b2[n * 16 + nlo];
    v8f acc;
    #pragma unroll
    for (int r = 0; r < 8; ++r) acc[r] = bv;
    #pragma unroll 1
    for (int kk = 0; kk < 16; ++kk)
      acc = wmma_bf16(load_a(a0 + kk * 32, HID, lane),
                      load_b(w2, HID, n * 16, kk * 32, lane), acc);
    #pragma unroll
    for (int r = 0; r < 8; ++r) a1[(hi8 + r) * HID + n * 16 + nlo] = f2bf(eluf_(acc[r]));
  }

  // Layer 3: a1 -> ELU -> a0
  #pragma unroll 1
  for (int n = 0; n < HID / 16; ++n) {
    float bv = b3[n * 16 + nlo];
    v8f acc;
    #pragma unroll
    for (int r = 0; r < 8; ++r) acc[r] = bv;
    #pragma unroll 1
    for (int kk = 0; kk < 16; ++kk)
      acc = wmma_bf16(load_a(a1 + kk * 32, HID, lane),
                      load_b(w3, HID, n * 16, kk * 32, lane), acc);
    #pragma unroll
    for (int r = 0; r < 8; ++r) a0[(hi8 + r) * HID + n * 16 + nlo] = f2bf(eluf_(acc[r]));
  }

  // mu head: a0 (16x512) -> out (16x64, f32)
  #pragma unroll 1
  for (int n = 0; n < OUTD / 16; ++n) {
    float bv = bmu[n * 16 + nlo];
    v8f acc;
    #pragma unroll
    for (int r = 0; r < 8; ++r) acc[r] = bv;
    #pragma unroll 1
    for (int kk = 0; kk < 16; ++kk)
      acc = wmma_bf16(load_a(a0 + kk * 32, HID, lane),
                      load_b(wmu, HID, n * 16, kk * 32, lane), acc);
    #pragma unroll
    for (int r = 0; r < 8; ++r) out[(rowbase + hi8 + r) * OUTD + n * 16 + nlo] = acc[r];
  }
}

// ---------------------------------------------------------------- launcher
extern "C" void kernel_launch(void* const* d_in, const int* in_sizes, int n_in,
                              void* d_out, int out_size, void* d_ws, size_t ws_size,
                              hipStream_t stream) {
  (void)in_sizes; (void)n_in; (void)out_size; (void)ws_size;
  const float* z_t  = (const float*)d_in[0];
  const float* z_g  = (const float*)d_in[1];
  const float* wemb = (const float*)d_in[2];
  const float* whh  = (const float*)d_in[3];
  const float* bih  = (const float*)d_in[4];
  const float* bhh  = (const float*)d_in[5];
  const float* w1   = (const float*)d_in[6];
  const float* b1   = (const float*)d_in[7];
  const float* w2   = (const float*)d_in[8];
  const float* b2   = (const float*)d_in[9];
  const float* w3   = (const float*)d_in[10];
  const float* b3   = (const float*)d_in[11];
  const float* wmu  = (const float*)d_in[12];
  const float* bmu  = (const float*)d_in[13];
  float* out = (float*)d_out;

  char* ws = (char*)d_ws;
  size_t off = 0;
  auto take = [&](size_t bytes) { char* p = ws + off; off = (off + bytes + 255) & ~(size_t)255; return p; };
  __bf16* whh_bf = (__bf16*)take(768 * 256 * 2);
  __bf16* w1_bf  = (__bf16*)take(512 * 256 * 2);
  __bf16* w2_bf  = (__bf16*)take(512 * 512 * 2);
  __bf16* w3_bf  = (__bf16*)take(512 * 512 * 2);
  __bf16* wmu_bf = (__bf16*)take(64 * 512 * 2);
  float*  h0     = (float*) take(512 * 256 * 4);
  __bf16* hs     = (__bf16*)take((size_t)512 * 128 * 256 * 2);

  k_convert<<<(768 * 256 + 255) / 256, 256, 0, stream>>>(whh, whh_bf, 768 * 256);
  k_convert<<<(512 * 256 + 255) / 256, 256, 0, stream>>>(w1,  w1_bf,  512 * 256);
  k_convert<<<(512 * 512 + 255) / 256, 256, 0, stream>>>(w2,  w2_bf,  512 * 512);
  k_convert<<<(512 * 512 + 255) / 256, 256, 0, stream>>>(w3,  w3_bf,  512 * 512);
  k_convert<<<(64 * 512 + 255) / 256, 256, 0, stream>>>(wmu, wmu_bf, 64 * 512);
  k_embed<<<BWROWS, 256, 0, stream>>>(z_t, z_g, wemb, h0);
  k_gru<<<32, 256, 0, stream>>>(h0, whh_bf, bih, bhh, hs);
  k_mlp<<<512, 256, 8 * 2 * 16 * HID * 2, stream>>>(hs, w1_bf, b1, w2_bf, b2, w3_bf, b3,
                                                    wmu_bf, bmu, out);
}
